// PolyAttn_31018253812234
// MI455X (gfx1250) — matrix operand order repaired and checked by reading, not compiled, not run on hardware
//
#include <hip/hip_runtime.h>
#include <hip/hip_bf16.h>

// ---------------------------------------------------------------------------
// PolyAttn for MI455X (gfx1250): bf16 WMMA GEMM pipeline with async
// global->LDS double buffering (GLOBAL_LOAD_ASYNC_TO_LDS_B128 / ASYNCcnt).
// ---------------------------------------------------------------------------

typedef __attribute__((ext_vector_type(16))) __bf16 v16bf;
typedef __attribute__((ext_vector_type(8)))  float  v8f;

struct Pack32 { uint4 lo, hi; };

static __device__ inline v16bf frag_from(uint4 lo, uint4 hi) {
    Pack32 p; p.lo = lo; p.hi = hi;
    return __builtin_bit_cast(v16bf, p);
}

#define BM 128
#define BN 128
#define BK 32
#define LDW 40   // row stride in elements (80 B) -> keeps 16B alignment, depads banks

// Async copy of 32 contiguous bytes global->LDS per lane (two b128 ops).
// ISA 10.x: INST_OFFSET is added to BOTH the LDS and the global address.
static __device__ inline void async_copy32(unsigned ldsOff, const void* gaddr) {
    asm volatile("global_load_async_to_lds_b128 %0, %1, off"
                 :: "v"(ldsOff), "v"((unsigned long long)gaddr) : "memory");
    asm volatile("global_load_async_to_lds_b128 %0, %1, off offset:16"
                 :: "v"(ldsOff), "v"((unsigned long long)gaddr) : "memory");
}

static __device__ inline void wait_async0() {
    asm volatile("s_wait_asynccnt 0x0" ::: "memory");
}

// One 32-deep K-step: 2 A frags x 4 B frags -> 8 WMMAs.
static __device__ inline void tile_mac(const __hip_bfloat16 (*As)[LDW],
                                       const __hip_bfloat16 (*Bs)[LDW],
                                       v8f acc[2][4],
                                       int m0, int n0, int l15, int hiA, int hiB)
{
    v16bf afr[2];
#pragma unroll
    for (int i = 0; i < 2; ++i) {
        const __hip_bfloat16* rp = &As[m0 + i * 16 + l15][0];
        afr[i] = frag_from(*(const uint4*)(rp + hiA),
                           *(const uint4*)(rp + hiA + 16));
    }
    v16bf bfr[4];
#pragma unroll
    for (int j = 0; j < 4; ++j) {
        const __hip_bfloat16* cp = &Bs[n0 + j * 16 + l15][0];
        bfr[j] = frag_from(*(const uint4*)(cp + hiB),
                           *(const uint4*)(cp + hiB + 16));
    }
#pragma unroll
    for (int i = 0; i < 2; ++i)
#pragma unroll
        for (int j = 0; j < 4; ++j)
            acc[i][j] = __builtin_amdgcn_wmma_f32_16x16x32_bf16(
                false, afr[i], false, bfr[j], (short)0, acc[i][j], false, false);
}

// C = act(A[M x Kdim] * Bt^T + bias), A row-major (strided), Bt stored [N][Kdim]
// (output-major, K-contiguous) so each lane's B fragment is two 16B runs of one row.
__global__ void __launch_bounds__(256) poly_gemm_bf16(
    const __hip_bfloat16* __restrict__ A, long long aBatchOff, long long aStride,
    const __hip_bfloat16* __restrict__ Bt, long long bBatchStride,
    const float* __restrict__ bias, long long biasBatchStride,
    __hip_bfloat16* __restrict__ C, long long cBatchOff, long long cStride,
    int Kdim, int doRelu)
{
    __shared__ __align__(16) __hip_bfloat16 As0[BM][LDW];
    __shared__ __align__(16) __hip_bfloat16 Bs0[BN][LDW];
    __shared__ __align__(16) __hip_bfloat16 As1[BM][LDW];
    __shared__ __align__(16) __hip_bfloat16 Bs1[BN][LDW];

    const int z = blockIdx.z;
    A  += (long long)z * aBatchOff;
    Bt += (long long)z * bBatchStride;
    C  += (long long)z * cBatchOff;
    if (bias) bias += (long long)z * biasBatchStride;

    const int tid   = threadIdx.x;
    const int lane  = tid & 31;
    const int wid   = tid >> 5;
    const int l15   = lane & 15;
    const int m0    = (wid >> 1) * 32;   // wave tile origin in M (4 waves)
    const int n0    = (wid & 1) * 64;    // wave tile origin in N (2 waves)
    const int rBase = blockIdx.y * BM;
    const int cBase = blockIdx.x * BN;

    const int cpRow  = tid >> 1;         // 0..127 : row of the LDS tile
    const int cpHalf = (tid & 1) * 16;   // element offset (0 or 16) = 32B half

    // Per-thread source rows (advance by k0 along K).
    const __hip_bfloat16* aRow = A  + (long long)(rBase + cpRow) * aStride + cpHalf;
    const __hip_bfloat16* bRow = Bt + (long long)(cBase + cpRow) * Kdim    + cpHalf;

    // LDS byte offsets for this thread's 32B slot in each buffer
    // (flat shared address: low 32 bits are the LDS offset).
    const unsigned ldsA0 = (unsigned)(unsigned long long)&As0[cpRow][cpHalf];
    const unsigned ldsB0 = (unsigned)(unsigned long long)&Bs0[cpRow][cpHalf];
    const unsigned ldsA1 = (unsigned)(unsigned long long)&As1[cpRow][cpHalf];
    const unsigned ldsB1 = (unsigned)(unsigned long long)&Bs1[cpRow][cpHalf];

    v8f acc[2][4] = {};

    // ISA 7.12.2: 16-bit A 16x32 — lanes<16 hold K {0..7,16..23}, lanes>=16
    // hold K {8..15,24..31}; 16-bit B 32x16 — same K order as A.
    const int hiA = (lane < 16) ? 0 : 8;
    const int hiB = (lane < 16) ? 0 : 8;

    // Prologue: fill buffer 0.
    async_copy32(ldsA0, aRow);
    async_copy32(ldsB0, bRow);
    wait_async0();
    __syncthreads();

    // Kdim is always a multiple of 2*BK here (1024 or 2048): unroll ping-pong.
    for (int k0 = 0; k0 < Kdim; k0 += 2 * BK) {
        // Prefetch tile k0+BK into buffer 1 while computing buffer 0.
        async_copy32(ldsA1, aRow + (k0 + BK));
        async_copy32(ldsB1, bRow + (k0 + BK));
        tile_mac(As0, Bs0, acc, m0, n0, l15, hiA, hiB);
        wait_async0();
        __syncthreads();

        // Prefetch tile k0+2*BK into buffer 0 while computing buffer 1.
        if (k0 + 2 * BK < Kdim) {
            async_copy32(ldsA0, aRow + (k0 + 2 * BK));
            async_copy32(ldsB0, bRow + (k0 + 2 * BK));
        }
        tile_mac(As1, Bs1, acc, m0, n0, l15, hiA, hiB);
        wait_async0();
        __syncthreads();
    }

    // Epilogue: C/D layout — vgpr e holds M = e (+8 for lanes 16..31), N = lane%16
    const int rowHi = (lane < 16) ? 0 : 8;
#pragma unroll
    for (int i = 0; i < 2; ++i) {
#pragma unroll
        for (int j = 0; j < 4; ++j) {
            const int col = cBase + n0 + j * 16 + l15;
            const float bv = bias ? bias[col] : 0.f;
            const int rTop = rBase + m0 + i * 16 + rowHi;
#pragma unroll
            for (int e = 0; e < 8; ++e) {
                float v = acc[i][j][e] + bv;
                if (doRelu) v = fmaxf(v, 0.f);
                C[(long long)(rTop + e) * cStride + col] = __float2bfloat16(v);
            }
        }
    }
}

// f32 [R][C] (batched) -> bf16 [C][R]  (weights to output-major, K-contiguous)
__global__ void poly_transpose_bf16(const float* __restrict__ in,
                                    __hip_bfloat16* __restrict__ out,
                                    int R, int Cc)
{
    const long long batch = blockIdx.z;
    in  += batch * (long long)R * Cc;
    out += batch * (long long)R * Cc;
    __shared__ float tile[32][33];
    const int c0 = blockIdx.x * 32, r0 = blockIdx.y * 32;
    for (int i = threadIdx.y; i < 32; i += 8)
        tile[i][threadIdx.x] = in[(long long)(r0 + i) * Cc + c0 + threadIdx.x];
    __syncthreads();
    for (int i = threadIdx.y; i < 32; i += 8)
        out[(long long)(c0 + i) * R + r0 + threadIdx.x] =
            __float2bfloat16(tile[threadIdx.x][i]);
}

// LayerNorm over H=1024, fp32 in, bf16 out (xn doubles as attention V)
__global__ void __launch_bounds__(256) poly_ln_bf16(
    const float* __restrict__ src, const float* __restrict__ g,
    const float* __restrict__ bta, __hip_bfloat16* __restrict__ xn)
{
    const long long tok = blockIdx.x;
    const float* x = src + tok * 1024;
    const int t = threadIdx.x;
    float v[4]; float s = 0.f, sq = 0.f;
#pragma unroll
    for (int e = 0; e < 4; ++e) { v[e] = x[t + e * 256]; s += v[e]; sq += v[e] * v[e]; }
#pragma unroll
    for (int o = 16; o > 0; o >>= 1) {
        s  += __shfl_xor(s,  o, 32);
        sq += __shfl_xor(sq, o, 32);
    }
    __shared__ float red[16];
    if ((t & 31) == 0) { red[t >> 5] = s; red[(t >> 5) + 8] = sq; }
    __syncthreads();
    if (t == 0) {
        float S = 0.f, Q = 0.f;
        for (int i = 0; i < 8; ++i) { S += red[i]; Q += red[i + 8]; }
        const float mu = S * (1.f / 1024.f);
        red[0] = mu;
        red[8] = Q * (1.f / 1024.f) - mu * mu;
    }
    __syncthreads();
    const float mu = red[0];
    const float rs = rsqrtf(red[8] + 1e-5f);
#pragma unroll
    for (int e = 0; e < 4; ++e) {
        const int c = t + e * 256;
        xn[tok * 1024 + c] = __float2bfloat16((v[e] - mu) * rs * g[c] + bta[c]);
    }
}

// Per-(b,h) 11x11 tanh attention + residual; bandwidth-trivial.
__global__ void __launch_bounds__(128) poly_attn_small(
    const __hip_bfloat16* __restrict__ q, const __hip_bfloat16* __restrict__ kk,
    const __hip_bfloat16* __restrict__ xn, const float* __restrict__ src,
    const float* __restrict__ bias_scale, float* __restrict__ out)
{
    const int b = blockIdx.x >> 3;
    const int h = blockIdx.x & 7;
    const int d = threadIdx.x;             // 0..127 == DH
    __shared__ float qs[11][128], ks[11][128], vs[11][128], ss[121];
    const long long base = ((long long)b * 11) * 1024 + h * 128 + d;
#pragma unroll
    for (int i = 0; i < 11; ++i) {
        const long long idx = base + (long long)i * 1024;
        qs[i][d] = __bfloat162float(q[idx]);
        ks[i][d] = __bfloat162float(kk[idx]);
        vs[i][d] = __bfloat162float(xn[idx]);
    }
    __syncthreads();
    if (d < 121) {
        const int i = d / 11, j = d % 11;
        float s = 0.f;
        for (int t = 0; t < 128; ++t) s += qs[i][t] * ks[j][t];
        s = tanhf(s * 0.08838834764831845f);        // 1/sqrt(128)
        s *= bias_scale[h * 11 + j] / (float)(j + 1);
        ss[d] = s;
    }
    __syncthreads();
#pragma unroll
    for (int i = 0; i < 11; ++i) {
        float acc = 0.f;
#pragma unroll
        for (int j = 0; j < 11; ++j) acc += ss[i * 11 + j] * vs[j][d];
        const long long idx = base + (long long)i * 1024;
        out[idx] = acc + src[idx];
    }
}

extern "C" void kernel_launch(void* const* d_in, const int* in_sizes, int n_in,
                              void* d_out, int out_size, void* d_ws, size_t ws_size,
                              hipStream_t stream)
{
    const float* src  = (const float*)d_in[0];
    const float* ln_g = (const float*)d_in[1];
    const float* ln_b = (const float*)d_in[2];
    const float* W1   = (const float*)d_in[3];
    const float* b1   = (const float*)d_in[4];
    const float* W2   = (const float*)d_in[5];
    const float* b2   = (const float*)d_in[6];
    const float* WQ   = (const float*)d_in[7];
    const float* WK   = (const float*)d_in[8];
    const float* bsc  = (const float*)d_in[9];
    float* out = (float*)d_out;

    const long long Bn = 16384, Kp = 11, Hd = 1024, HM = 2048, NH = 8;

    char* ws = (char*)d_ws;
    size_t off = 0;
    auto take = [&](size_t bytes) -> char* {
        char* p = ws + off;
        off += (bytes + 255) & ~(size_t)255;
        return p;
    };

    __hip_bfloat16* xn   = (__hip_bfloat16*)take((size_t)Bn * Kp * Hd * 2);
    __hip_bfloat16* tbuf = (__hip_bfloat16*)take((size_t)Bn * Kp * HM * 2);
    __hip_bfloat16* tok  = (__hip_bfloat16*)take((size_t)Bn * Kp * Hd * 2);
    __hip_bfloat16* W1t  = (__hip_bfloat16*)take((size_t)Kp * Hd * HM * 2);
    __hip_bfloat16* W2t  = (__hip_bfloat16*)take((size_t)Kp * Hd * HM * 2);
    __hip_bfloat16* WQt  = (__hip_bfloat16*)take((size_t)Hd * Hd * 2);
    __hip_bfloat16* WKt  = (__hip_bfloat16*)take((size_t)Hd * Hd * 2);
    // q/k reuse tbuf: HM == 2*Hd so the t region holds both exactly.
    __hip_bfloat16* qb = tbuf;
    __hip_bfloat16* kb = tbuf + (size_t)Bn * Kp * Hd;

    // Weights -> bf16, output-major [N][Kdim]
    poly_transpose_bf16<<<dim3(HM / 32, Hd / 32, Kp), dim3(32, 8), 0, stream>>>(W1, W1t, (int)Hd, (int)HM);
    poly_transpose_bf16<<<dim3(Hd / 32, HM / 32, Kp), dim3(32, 8), 0, stream>>>(W2, W2t, (int)HM, (int)Hd);
    poly_transpose_bf16<<<dim3(Hd / 32, Hd / 32, 1),  dim3(32, 8), 0, stream>>>(WQ, WQt, (int)Hd, (int)Hd);
    poly_transpose_bf16<<<dim3(Hd / 32, Hd / 32, 1),  dim3(32, 8), 0, stream>>>(WK, WKt, (int)Hd, (int)Hd);

    // LayerNorm -> xn (bf16)
    poly_ln_bf16<<<dim3((unsigned)(Bn * Kp)), dim3(256), 0, stream>>>(src, ln_g, ln_b, xn);

    // GEMM1 (per expert k): t = relu(xn @ W1 + b1)   [16384 x 1024] x [1024 x 2048]
    poly_gemm_bf16<<<dim3(HM / 128, Bn / 128, Kp), dim3(256), 0, stream>>>(
        xn, Hd, Kp * Hd, W1t, Hd * HM, b1, HM, tbuf, HM, Kp * HM, (int)Hd, 1);
    // GEMM2 (per expert k): token = t @ W2 + b2      [16384 x 2048] x [2048 x 1024]
    poly_gemm_bf16<<<dim3(Hd / 128, Bn / 128, Kp), dim3(256), 0, stream>>>(
        tbuf, HM, Kp * HM, W2t, Hd * HM, b2, Hd, tok, Hd, Kp * Hd, (int)HM, 0);
    // GEMM3: q = token @ WQ ; k = token @ WK         [180224 x 1024] x [1024 x 1024]
    poly_gemm_bf16<<<dim3(Hd / 128, (Bn * Kp) / 128, 1), dim3(256), 0, stream>>>(
        tok, 0, Hd, WQt, 0, nullptr, 0, qb, 0, Hd, (int)Hd, 0);
    poly_gemm_bf16<<<dim3(Hd / 128, (Bn * Kp) / 128, 1), dim3(256), 0, stream>>>(
        tok, 0, Hd, WKt, 0, nullptr, 0, kb, 0, Hd, (int)Hd, 0);

    // Attention (tiny) + residual, fp32 out
    poly_attn_small<<<dim3((unsigned)(Bn * NH)), dim3(128), 0, stream>>>(
        qb, kb, xn, src, bsc, out);

    (void)in_sizes; (void)n_in; (void)out_size; (void)ws_size;
}
